// CausalSelfAttention_43009802502260
// MI455X (gfx1250) — compile-verified, matrix operand order checked by reading, not run on hardware
//
#include <hip/hip_runtime.h>
#include <hip/hip_bf16.h>

// ---------------------------------------------------------------------------
// Problem constants (match reference)
// ---------------------------------------------------------------------------
constexpr int Bb  = 2;
constexpr int Tt  = 2048;
constexpr int Cc  = 1024;
constexpr int Hh  = 16;
constexpr int MM  = 256;           // external memory rows
constexpr int Ss  = MM + Tt;       // 2304 keys per head
constexpr int HDd = 64;            // head dim
constexpr float SCALE = 0.125f;    // 1/sqrt(64)
constexpr int QB  = 128;           // queries per attention workgroup (8 waves)

typedef __attribute__((ext_vector_type(16))) _Float16 v16h;
typedef __attribute__((ext_vector_type(8)))  float    v8f;

// ---------------------------------------------------------------------------
// WMMA helpers (CDNA5 wave32, V_WMMA_F32_16X16X32_F16)
//
// 16-bit A operand layout (ISA 7.12.2): lane L -> row m = L%16, hi = L/16.
// Per lane the 16 halves are two contiguous 16B chunks of the 32-wide K slice:
//   halves [8*hi, 8*hi+8)  and  [16+8*hi, 16+8*hi+8)
// B operand mirrors this with lane -> column n, so both fragments load from
// K-contiguous row-major storage with two b128 loads.
// ---------------------------------------------------------------------------
__device__ __forceinline__ v16h load_frag(const _Float16* base, int hi) {
    union { int4 i[2]; v16h h; } u;
    const int4* p = (const int4*)base;
    u.i[0] = p[hi];
    u.i[1] = p[2 + hi];
    return u.h;
}

__device__ __forceinline__ v8f wmma_f16(v16h a, v16h b, v8f c) {
    return __builtin_amdgcn_wmma_f32_16x16x32_f16(
        /*neg_a=*/false, a, /*neg_b=*/false, b,
        /*c_mod=*/(short)0, c, /*reuse_a=*/false, /*reuse_b=*/false);
}

// ---------------------------------------------------------------------------
// Elementwise f32 -> f16 convert
// ---------------------------------------------------------------------------
__global__ void cvt_f16_kernel(const float* __restrict__ src,
                               _Float16* __restrict__ dst, int n) {
    int i = blockIdx.x * blockDim.x + threadIdx.x;
    if (i < n) dst[i] = (_Float16)src[i];
}

// ---------------------------------------------------------------------------
// Transposing convert: W [K][N] f32 -> Wt [N][K] f16 (LDS tiled)
// ---------------------------------------------------------------------------
__global__ __launch_bounds__(256)
void transpose_cvt_kernel(const float* __restrict__ W,
                          _Float16* __restrict__ Wt, int K, int N) {
    __shared__ float tile[32][33];
    int kb = blockIdx.x * 32;
    int nb = blockIdx.y * 32;
    int tx = threadIdx.x;          // 0..31
    int ty = threadIdx.y;          // 0..7
    #pragma unroll
    for (int i = 0; i < 32; i += 8)
        tile[ty + i][tx] = W[(size_t)(kb + ty + i) * N + nb + tx];
    __syncthreads();
    #pragma unroll
    for (int i = 0; i < 32; i += 8)
        Wt[(size_t)(nb + ty + i) * K + kb + tx] = (_Float16)tile[tx][ty + i];
}

// ---------------------------------------------------------------------------
// ext_mem prefix -> K/V head-major buffers (rows 0..MEM-1), f16
// Kh/Vh layout: [B][H][Ss][64]
// ---------------------------------------------------------------------------
__global__ void mem_prefix_kernel(const float* __restrict__ em,
                                  _Float16* __restrict__ Kh,
                                  _Float16* __restrict__ Vh) {
    int i = blockIdx.x * blockDim.x + threadIdx.x;
    if (i >= Bb * MM * Cc) return;
    int c   = i % Cc;
    int rem = i / Cc;
    int s   = rem % MM;
    int b   = rem / MM;
    int h   = c / HDd;
    int d   = c % HDd;
    _Float16 v = (_Float16)em[i];
    size_t idx = (((size_t)b * Hh + h) * Ss + s) * HDd + d;
    Kh[idx] = v;
    Vh[idx] = v;
}

// ---------------------------------------------------------------------------
// WMMA GEMM:  C[M][N] = A[M][K] * Bt[N][K]^T + bias
// Block 128x128, 256 threads = 8 waves (2x4), wave tile 64x32 (4x2 subtiles).
// Register double-buffered global->LDS pipeline + global_prefetch_b8 two
// K-slices ahead.
// EPI==0: scatter into Qh/Kh/Vh f16 head buffers (qkv epilogue)
// EPI==1: f32 output + bias (projection epilogue)
// ---------------------------------------------------------------------------
template <int EPI>
__global__ __launch_bounds__(256)
void gemm_wmma_kernel(const _Float16* __restrict__ A,
                      const _Float16* __restrict__ Bt,
                      const float* __restrict__ bias,
                      _Float16* __restrict__ q_out,
                      _Float16* __restrict__ k_out,
                      _Float16* __restrict__ v_out,
                      float* __restrict__ f_out,
                      int M, int N, int K) {
    __shared__ __align__(16) _Float16 As[128 * 32];
    __shared__ __align__(16) _Float16 Bs[128 * 32];

    const int tid  = threadIdx.x;
    const int lane = tid & 31;
    const int wave = tid >> 5;
    const int wm   = wave >> 2;        // 0..1
    const int wn   = wave & 3;         // 0..3
    const int hi   = (lane >> 4) & 1;
    const int ln   = lane & 15;

    const int m0 = blockIdx.x * 128;
    const int n0 = blockIdx.y * 128;

    // This thread's two int4 slots in the 128x32 tiles.
    const int c0   = tid * 2;              // 0..510 step 2
    const int row0 = (c0)     >> 2, qq0 = (c0)     & 3;
    const int row1 = (c0 + 1) >> 2, qq1 = (c0 + 1) & 3;
    const _Float16* Arow0 = A  + (size_t)(m0 + row0) * K;
    const _Float16* Arow1 = A  + (size_t)(m0 + row1) * K;
    const _Float16* Brow0 = Bt + (size_t)(n0 + row0) * K;
    const _Float16* Brow1 = Bt + (size_t)(n0 + row1) * K;

    v8f acc[4][2] = {};
    int4 ra[2], rb[2];

    // Prime the register pipeline with K-slice 0.
    ra[0] = *((const int4*)(Arow0) + qq0);
    ra[1] = *((const int4*)(Arow1) + qq1);
    rb[0] = *((const int4*)(Brow0) + qq0);
    rb[1] = *((const int4*)(Brow1) + qq1);

    for (int k0 = 0; k0 < K; k0 += 32) {
        __syncthreads();                       // LDS reads of prev slice done
        ((int4*)As)[c0]     = ra[0];
        ((int4*)As)[c0 + 1] = ra[1];
        ((int4*)Bs)[c0]     = rb[0];
        ((int4*)Bs)[c0 + 1] = rb[1];
        __syncthreads();

        // Issue next slice's global loads; they complete under the WMMAs.
        if (k0 + 32 < K) {
            ra[0] = *((const int4*)(Arow0 + k0 + 32) + qq0);
            ra[1] = *((const int4*)(Arow1 + k0 + 32) + qq1);
            rb[0] = *((const int4*)(Brow0 + k0 + 32) + qq0);
            rb[1] = *((const int4*)(Brow1 + k0 + 32) + qq1);
        }
        // Prefetch two slices ahead (global_prefetch_b8).
        if (k0 + 64 < K) {
            __builtin_prefetch(Arow0 + k0 + 64, 0, 3);
            __builtin_prefetch(Brow0 + k0 + 64, 0, 3);
        }

        v16h af[4], bf[2];
        #pragma unroll
        for (int mi = 0; mi < 4; ++mi)
            af[mi] = load_frag(&As[(wm * 64 + mi * 16 + ln) * 32], hi);
        #pragma unroll
        for (int ni = 0; ni < 2; ++ni)
            bf[ni] = load_frag(&Bs[(wn * 32 + ni * 16 + ln) * 32], hi);

        #pragma unroll
        for (int mi = 0; mi < 4; ++mi)
            #pragma unroll
            for (int ni = 0; ni < 2; ++ni)
                acc[mi][ni] = wmma_f16(af[mi], bf[ni], acc[mi][ni]);
    }

    // Epilogue. C/D layout: vgpr r, lane -> (m = 8*hi + r, n = ln) within tile.
    #pragma unroll
    for (int mi = 0; mi < 4; ++mi) {
        #pragma unroll
        for (int ni = 0; ni < 2; ++ni) {
            int col  = n0 + wn * 32 + ni * 16 + ln;
            float bv = bias[col];
            #pragma unroll
            for (int r = 0; r < 8; ++r) {
                int row   = m0 + wm * 64 + mi * 16 + hi * 8 + r;
                float val = acc[mi][ni][r] + bv;
                if constexpr (EPI == 0) {
                    int which = col >> 10;          // 0=q 1=k 2=v
                    int c     = col & (Cc - 1);
                    int hh    = c >> 6;
                    int d     = c & 63;
                    int b     = row / Tt;
                    int t     = row % Tt;
                    if (which == 0)
                        q_out[(((size_t)b * Hh + hh) * Tt + t) * HDd + d] = (_Float16)val;
                    else if (which == 1)
                        k_out[(((size_t)b * Hh + hh) * Ss + (MM + t)) * HDd + d] = (_Float16)val;
                    else
                        v_out[(((size_t)b * Hh + hh) * Ss + (MM + t)) * HDd + d] = (_Float16)val;
                } else {
                    f_out[(size_t)row * N + col] = val;
                }
            }
        }
    }
}

// ---------------------------------------------------------------------------
// Flash attention, one (b,h) x QB-query block per workgroup.
// 256 threads = 8 waves; each wave owns 16 query rows.
// Key tiles of 64 in LDS (K row-major, V transposed). Online softmax.
// Next K/V tile is prefetched (global_prefetch_b8) while computing.
// Yh layout: [B][T][C] f16 (row-major activations for the proj GEMM).
// ---------------------------------------------------------------------------
__global__ __launch_bounds__(256)
void attn_kernel(const _Float16* __restrict__ Qh,
                 const _Float16* __restrict__ Kh,
                 const _Float16* __restrict__ Vh,
                 _Float16* __restrict__ Yh) {
    __shared__ __align__(16) _Float16 Ks[64 * 64];     // [key][d]
    __shared__ __align__(16) _Float16 Vt[64 * 64];     // [d][key]
    __shared__ __align__(16) _Float16 Ps[8][16 * 64];  // per-wave P [q][key]

    const int tid  = threadIdx.x;
    const int lane = tid & 31;
    const int wave = tid >> 5;
    const int hi   = (lane >> 4) & 1;
    const int ln   = lane & 15;

    const int bh = blockIdx.x;            // b*Hh + h
    const int qt = blockIdx.y;            // query block of QB
    const int b  = bh / Hh;
    const int h  = bh % Hh;
    const int q0 = qt * QB + wave * 16;   // this wave's first query row

    const _Float16* Qbh = Qh + (size_t)bh * Tt * HDd;
    const _Float16* Kbh = Kh + (size_t)bh * Ss * HDd;
    const _Float16* Vbh = Vh + (size_t)bh * Ss * HDd;

    // Q fragments (constant over the key loop): depth 0..31 and 32..63
    const _Float16* qrow = Qbh + (size_t)(q0 + ln) * HDd;
    v16h aq0 = load_frag(qrow,       hi);
    v16h aq1 = load_frag(qrow + 32,  hi);

    const float NEG_INF = -__builtin_inff();
    float mrun[8], lrun[8];
    v8f   yacc[4] = {};
    #pragma unroll
    for (int r = 0; r < 8; ++r) { mrun[r] = NEG_INF; lrun[r] = 0.0f; }

    const int tg_base = q0 + hi * 8;              // query row = tg_base + r
    const int s_end   = MM + qt * QB + QB;        // keys needed by this block

    for (int s0 = 0; s0 < s_end; s0 += 64) {
        __syncthreads();
        // K tile: 8 KB, fully contiguous in global memory for this (b,h).
        #pragma unroll
        for (int i = 0; i < 2; ++i) {
            int ci = tid + i * 256;               // 0..511 int4
            ((int4*)Ks)[ci] = ((const int4*)(Kbh + (size_t)s0 * HDd))[ci];
        }
        // V tile, transposed into Vt[d][key]. Thread: 16 d-values of one key.
        {
            int sl = tid & 63;
            int d0 = (tid >> 6) * 16;             // 0,16,32,48
            const int4* vr = (const int4*)(Vbh + (size_t)(s0 + sl) * HDd + d0);
            #pragma unroll
            for (int c2 = 0; c2 < 2; ++c2) {
                union { int4 v; _Float16 e[8]; } u;
                u.v = vr[c2];
                #pragma unroll
                for (int e = 0; e < 8; ++e)
                    Vt[(d0 + c2 * 8 + e) * 64 + sl] = u.e[e];
            }
        }
        // Prefetch next tile: one 128B line per thread covers both 8KB tiles.
        {
            int s1 = s0 + 64;
            if (s1 + 64 <= Ss) {
                const _Float16* base = (tid < 128) ? Kbh : Vbh;
                __builtin_prefetch(base + (size_t)s1 * HDd + (size_t)(tid & 127) * 64, 0, 3);
            }
        }
        __syncthreads();

        // ---- scores: 4 subtiles of 16 keys, depth 64 = 2 WMMAs each ----
        v8f sc[4];
        #pragma unroll
        for (int kt = 0; kt < 4; ++kt) {
            v16h bk0 = load_frag(&Ks[(kt * 16 + ln) * 64],      hi);
            v16h bk1 = load_frag(&Ks[(kt * 16 + ln) * 64 + 32], hi);
            v8f c = {};
            c = wmma_f16(aq0, bk0, c);
            c = wmma_f16(aq1, bk1, c);
            sc[kt] = c;
        }

        // ---- scale + mask, per-row tile max ----
        float tmax[8];
        #pragma unroll
        for (int r = 0; r < 8; ++r) tmax[r] = NEG_INF;
        #pragma unroll
        for (int kt = 0; kt < 4; ++kt) {
            int sg = s0 + kt * 16 + ln;           // global key index
            #pragma unroll
            for (int r = 0; r < 8; ++r) {
                int tg  = tg_base + r;
                bool ok = (sg < MM) || ((sg - MM) <= tg);
                float v = ok ? sc[kt][r] * SCALE : NEG_INF;
                sc[kt][r] = v;
                tmax[r]   = fmaxf(tmax[r], v);
            }
        }
        #pragma unroll
        for (int r = 0; r < 8; ++r)
            #pragma unroll
            for (int off = 1; off < 16; off <<= 1)
                tmax[r] = fmaxf(tmax[r], __shfl_xor(tmax[r], off, 32));

        // ---- online softmax update ----
        float psum[8];
        #pragma unroll
        for (int r = 0; r < 8; ++r) {
            float mnew  = fmaxf(mrun[r], tmax[r]);
            float alpha = __expf(mrun[r] - mnew);
            mrun[r] = mnew;
            lrun[r] *= alpha;
            float s = 0.0f;
            #pragma unroll
            for (int kt = 0; kt < 4; ++kt) {
                float p = __expf(sc[kt][r] - mnew);
                sc[kt][r] = p;
                s += p;
            }
            psum[r] = s;
            #pragma unroll
            for (int ni = 0; ni < 4; ++ni) yacc[ni][r] *= alpha;
        }
        #pragma unroll
        for (int r = 0; r < 8; ++r) {
            #pragma unroll
            for (int off = 1; off < 16; off <<= 1)
                psum[r] += __shfl_xor(psum[r], off, 32);
            lrun[r] += psum[r];
        }

        // ---- P -> per-wave LDS (f16, row-major [16][64]) ----
        _Float16* Pw = &Ps[wave][0];
        #pragma unroll
        for (int kt = 0; kt < 4; ++kt)
            #pragma unroll
            for (int r = 0; r < 8; ++r)
                Pw[(hi * 8 + r) * 64 + kt * 16 + ln] = (_Float16)sc[kt][r];

        // ---- y += P * V : 2 key-depth chunks x 4 output subtiles ----
        #pragma unroll
        for (int kc = 0; kc < 2; ++kc) {
            v16h ap = load_frag(&Pw[ln * 64 + kc * 32], hi);
            #pragma unroll
            for (int ni = 0; ni < 4; ++ni) {
                v16h bv = load_frag(&Vt[(ni * 16 + ln) * 64 + kc * 32], hi);
                yacc[ni] = wmma_f16(ap, bv, yacc[ni]);
            }
        }
    }

    // ---- normalize and store Yh[b][t][h*64 + d] ----
    #pragma unroll
    for (int ni = 0; ni < 4; ++ni)
        #pragma unroll
        for (int r = 0; r < 8; ++r) {
            int tg  = tg_base + r;
            float v = yacc[ni][r] / lrun[r];
            Yh[((size_t)b * Tt + tg) * Cc + h * HDd + ni * 16 + ln] = (_Float16)v;
        }
}

// ---------------------------------------------------------------------------
// Host launcher.  Workspace (f16): Xh | WattnT | WprojT | Qh | Kh | Vh | Yh
// Total ~50 MB.
// ---------------------------------------------------------------------------
extern "C" void kernel_launch(void* const* d_in, const int* in_sizes, int n_in,
                              void* d_out, int out_size, void* d_ws, size_t ws_size,
                              hipStream_t stream) {
    (void)in_sizes; (void)n_in; (void)out_size; (void)ws_size;

    const float* x       = (const float*)d_in[0];
    const float* ext_mem = (const float*)d_in[1];
    const float* W_attn  = (const float*)d_in[2];
    const float* b_attn  = (const float*)d_in[3];
    const float* W_proj  = (const float*)d_in[4];
    const float* b_proj  = (const float*)d_in[5];
    float*       out     = (float*)d_out;

    _Float16* Xh     = (_Float16*)d_ws;
    _Float16* WattnT = Xh     + (size_t)Bb * Tt * Cc;        // 4096*1024
    _Float16* WprojT = WattnT + (size_t)3 * Cc * Cc;         // 3072*1024
    _Float16* Qh     = WprojT + (size_t)Cc * Cc;             // 1024*1024
    _Float16* Kh     = Qh     + (size_t)Bb * Hh * Tt * HDd;
    _Float16* Vh     = Kh     + (size_t)Bb * Hh * Ss * HDd;
    _Float16* Yh     = Vh     + (size_t)Bb * Hh * Ss * HDd;

    const int M = Bb * Tt;           // 4096

    // Stage 0: precision staging.
    {
        int n = Bb * Tt * Cc;
        cvt_f16_kernel<<<(n + 255) / 256, 256, 0, stream>>>(x, Xh, n);
    }
    transpose_cvt_kernel<<<dim3(Cc / 32, 3 * Cc / 32), dim3(32, 8), 0, stream>>>(
        W_attn, WattnT, Cc, 3 * Cc);
    transpose_cvt_kernel<<<dim3(Cc / 32, Cc / 32), dim3(32, 8), 0, stream>>>(
        W_proj, WprojT, Cc, Cc);
    {
        int n = Bb * MM * Cc;
        mem_prefix_kernel<<<(n + 255) / 256, 256, 0, stream>>>(ext_mem, Kh, Vh);
    }

    // Stage 1: QKV GEMM  [4096x1024] x [1024x3072], scatter epilogue.
    gemm_wmma_kernel<0><<<dim3(M / 128, (3 * Cc) / 128), 256, 0, stream>>>(
        Xh, WattnT, b_attn, Qh, Kh, Vh, nullptr, M, 3 * Cc, Cc);

    // Stage 2: flash attention (B*H blocks x T/QB query tiles).
    attn_kernel<<<dim3(Bb * Hh, Tt / QB), 256, 0, stream>>>(Qh, Kh, Vh, Yh);

    // Stage 3: projection GEMM  [4096x1024] x [1024x1024] -> f32 out.
    gemm_wmma_kernel<1><<<dim3(M / 128, Cc / 128), 256, 0, stream>>>(
        Yh, WprojT, b_proj, nullptr, nullptr, nullptr, out, M, Cc, Cc);
}
